// MulMamba_86930138071594
// MI455X (gfx1250) — matrix operand order repaired and checked by reading, not compiled, not run on hardware
//
#include <hip/hip_runtime.h>
#include <hip/hip_bf16.h>
#include <math.h>

#define DEVINL __device__ __forceinline__

namespace {

constexpr int Bn  = 2;
constexpr int Cc  = 96;
constexpr int Hh  = 40;
constexpr int Ww  = 40;
constexpr int Ls  = Hh * Ww;      // 1600
constexpr int DIN = 192;
constexpr int Ns  = 16;
constexpr int Rr  = 6;
constexpr int Kk  = 4;
constexpr int CP  = Rr + 2 * Ns;  // 38
constexpr int CPP = 48;           // padded to 3 WMMA tiles
constexpr int HID = DIN / 16;     // 12

typedef __bf16 bf16_t;
typedef __attribute__((ext_vector_type(16))) __bf16 v16bf;
typedef __attribute__((ext_vector_type(8)))  float  v8f;

// float -> bf16 round-to-nearest-even via bit manipulation (no reliance on
// native f32->bf16 conversion support).
DEVINL unsigned short bfbits(float f) {
  unsigned u = __float_as_uint(f);
  unsigned r = u + 0x7fffu + ((u >> 16) & 1u);
  return (unsigned short)(r >> 16);
}
DEVINL bf16_t tobf(float f) { return __builtin_bit_cast(bf16_t, bfbits(f)); }
DEVINL bf16_t rawbf(unsigned short s) { return __builtin_bit_cast(bf16_t, s); }

// One 16x16x32 bf16 WMMA step. Fragment layouts per CDNA5 ISA 7.12.2:
//  A 16x32: lanes 0-15 -> M=lane, elems 0..7 = K0..7, 8..15 = K16..23
//           lanes16-31 -> M=lane-16, elems 0..7 = K8..15, 8..15 = K24..31
//  B 32x16: lanes 0-15 -> N=lane, elems 0..7 = K0..7, 8..15 = K16..23 ; lanes16-31 -> K8..15, K24..31
//  C/D 16x16 f32: N = lane&15, M = v + 8*(lane>>4) for acc element v.
template <typename LA, typename LB>
DEVINL v8f wmma_step(int lane, LA la, LB lb, v8f acc) {
  v16bf a, b;
  const int mr = lane & 15;
  const int hi = (lane >> 4) & 1;
#pragma unroll
  for (int e = 0; e < 16; ++e) {
    int kk = ((e >> 3) << 4) | (hi << 3) | (e & 7);
    a[e] = la(mr, kk);
  }
#pragma unroll
  for (int e = 0; e < 16; ++e) {
    int kk = ((e >> 3) << 4) | (hi << 3) | (e & 7);
    b[e] = lb(kk, mr);
  }
  return __builtin_amdgcn_wmma_f32_16x16x32_bf16(false, a, false, b,
                                                 (short)0, acc, false, false);
}

DEVINL float silu_f(float x) { return x / (1.f + __expf(-x)); }
DEVINL float sigmoid_f(float x) { return 1.f / (1.f + __expf(-x)); }

// cross_scan element: xs[b,k,d,l] from activation (B,DIN,H,W)
DEVINL float load_xs(const float* __restrict__ act, int b, int k, int d, int l) {
  int ls = (k & 2) ? (Ls - 1 - l) : l;
  int s  = (k & 1) ? ((ls % Hh) * Ww + (ls / Hh)) : ls;
  return act[((size_t)b * DIN + d) * Ls + s];
}

// ---------------- Kernel 1: in_proj GEMM (bf16 WMMA) ----------------
// pre[b,o,l] = sum_c W[o,c] * X[b,c,l]; M=192, N=1600, K=96
__global__ __launch_bounds__(32)
void k_inproj(const float* __restrict__ xr, const float* __restrict__ xe,
              const float* __restrict__ wr, const float* __restrict__ we,
              float* __restrict__ pre_r, float* __restrict__ pre_e) {
  const int lane = threadIdx.x;
  const int tn = blockIdx.x, tm = blockIdx.y, z = blockIdx.z;
  const int b = z >> 1, mod = z & 1;
  const float* X  = mod ? xe : xr;
  const float* Wt = mod ? we : wr;
  float* O = (mod ? pre_e : pre_r) + (size_t)b * DIN * Ls;
  v8f acc = {};
  for (int kc = 0; kc < Cc; kc += 32) {
    acc = wmma_step(lane,
        [&](int m, int kk) { return tobf(Wt[(tm * 16 + m) * Cc + kc + kk]); },
        [&](int kk, int n) {
          return tobf(X[((size_t)b * Cc + kc + kk) * Ls + tn * 16 + n]);
        },
        acc);
  }
  const int nn = lane & 15, mb = ((lane >> 4) & 1) * 8;
#pragma unroll
  for (int v = 0; v < 8; ++v)
    O[(size_t)(tm * 16 + mb + v) * Ls + tn * 16 + nn] = acc[v];
}

// ---------------- Kernel 2: depthwise 3x3 + bias + SiLU ----------------
__global__ void k_conv(const float* __restrict__ pre_r, const float* __restrict__ pre_e,
                       const float* __restrict__ cw, const float* __restrict__ cb,
                       float* __restrict__ act_r, float* __restrict__ act_e) {
  int idx = blockIdx.x * blockDim.x + threadIdx.x;
  const int total = 2 * Bn * DIN * Ls;
  if (idx >= total) return;
  int s = idx % Ls; int t = idx / Ls;
  int d = t % DIN; t /= DIN;
  int b = t % Bn; int mod = t / Bn;
  const float* P = (mod ? pre_e : pre_r) + ((size_t)b * DIN + d) * Ls;
  int h = s / Ww, w = s % Ww;
  float sum = cb[d];
#pragma unroll
  for (int dh = -1; dh <= 1; ++dh)
#pragma unroll
    for (int dw = -1; dw <= 1; ++dw) {
      int hh = h + dh, w2 = w + dw;
      if (hh >= 0 && hh < Hh && w2 >= 0 && w2 < Ww)
        sum += cw[d * 9 + (dh + 1) * 3 + (dw + 1)] * P[hh * Ww + w2];
    }
  (mod ? act_e : act_r)[((size_t)b * DIN + d) * Ls + s] = silu_f(sum);
}

// ---------------- Kernel 3: x_proj GEMM (bf16 WMMA) ----------------
// T[b,k,c,l] = sum_d xpw[k,c,d] * xs[b,k,d,l]; M=38(pad48), N=1600, K=192
__global__ __launch_bounds__(32)
void k_xproj(const float* __restrict__ act_r, const float* __restrict__ act_e,
             const float* __restrict__ xw1, const float* __restrict__ xw2,
             float* __restrict__ T_r, float* __restrict__ T_e) {
  const int lane = threadIdx.x;
  const int tn = blockIdx.x, tm = blockIdx.y;
  int z = blockIdx.z;
  const int k = z % Kk; z /= Kk;
  const int b = z % Bn; const int mod = z / Bn;
  const float* act = mod ? act_e : act_r;
  const float* XW = (mod ? xw2 : xw1) + (size_t)k * CP * DIN;
  float* T = (mod ? T_e : T_r) + (size_t)(b * Kk + k) * CPP * Ls;
  v8f acc = {};
  for (int kc = 0; kc < DIN; kc += 32) {
    acc = wmma_step(lane,
        [&](int m, int kk) {
          int c = tm * 16 + m;
          return (c < CP) ? tobf(XW[(size_t)c * DIN + kc + kk]) : tobf(0.f);
        },
        [&](int kk, int n) {
          return tobf(load_xs(act, b, k, kc + kk, tn * 16 + n));
        },
        acc);
  }
  const int nn = lane & 15, mb = ((lane >> 4) & 1) * 8;
#pragma unroll
  for (int v = 0; v < 8; ++v)
    T[(size_t)(tm * 16 + mb + v) * Ls + tn * 16 + nn] = acc[v];
}

// ---------------- Kernel 4: selective scan ----------------
// One thread per (mod,b,k,d); N=16 state in registers; dt folded in from T.
__global__ void k_scan(const float* __restrict__ act_r, const float* __restrict__ act_e,
                       const float* __restrict__ T_r, const float* __restrict__ T_e,
                       const float* __restrict__ A1, const float* __restrict__ A2,
                       const float* __restrict__ dtw1, const float* __restrict__ dtb1,
                       const float* __restrict__ dtw2, const float* __restrict__ dtb2,
                       const float* __restrict__ D1, const float* __restrict__ D2,
                       float* __restrict__ y_r, float* __restrict__ y_e) {
  int idx = blockIdx.x * blockDim.x + threadIdx.x;
  const int total = 2 * Bn * Kk * DIN;
  if (idx >= total) return;
  int d = idx % DIN; int t = idx / DIN;
  int k = t % Kk; t /= Kk;
  int b = t % Bn; int mod = t / Bn;
  const float* act   = mod ? act_e : act_r;
  const float* Tself = (mod ? T_e : T_r) + (size_t)(b * Kk + k) * CPP * Ls;
  const float* Toth  = (mod ? T_r : T_e) + (size_t)(b * Kk + k) * CPP * Ls; // cross C
  const float* Alog  = mod ? A2 : A1;
  const float* dtw   = mod ? dtw2 : dtw1;
  const float* dtb   = mod ? dtb2 : dtb1;
  const float* Dv_   = mod ? D1 : D2;  // cross D: y_r uses D_2, y_e uses D_1
  float* Y = (mod ? y_e : y_r) + ((size_t)(b * Kk + k) * DIN + d) * Ls;
  const int kd = k * DIN + d;

  float Areg[Ns];
#pragma unroll
  for (int n = 0; n < Ns; ++n) Areg[n] = -__expf(Alog[kd * Ns + n]);
  float wreg[Rr];
#pragma unroll
  for (int r = 0; r < Rr; ++r) wreg[r] = dtw[kd * Rr + r];
  const float bias = dtb[kd];
  const float Dv = Dv_[kd];

  float hst[Ns];
#pragma unroll
  for (int n = 0; n < Ns; ++n) hst[n] = 0.f;

  for (int l = 0; l < Ls; ++l) {
    float dtr = bias;
#pragma unroll
    for (int r = 0; r < Rr; ++r) dtr += wreg[r] * Tself[r * Ls + l];
    float dt = (dtr > 20.f) ? dtr : log1pf(__expf(dtr));  // softplus
    float u = load_xs(act, b, k, d, l);
    float dtu = dt * u;
    float y = 0.f;
#pragma unroll
    for (int n = 0; n < Ns; ++n) {
      float dA = __expf(dt * Areg[n]);
      hst[n] = dA * hst[n] + dtu * Tself[(Rr + n) * Ls + l];
      y += hst[n] * Toth[(Rr + Ns + n) * Ls + l];
    }
    Y[l] = y + Dv * u;
  }
}

// ---------------- Kernel 5: cross_merge (deterministic gather) ----------------
__global__ void k_merge(const float* __restrict__ y_r, const float* __restrict__ y_e,
                        float* __restrict__ ym_r, float* __restrict__ ym_e) {
  int idx = blockIdx.x * blockDim.x + threadIdx.x;
  const int total = 2 * Bn * DIN * Ls;
  if (idx >= total) return;
  int s = idx % Ls; int t = idx / Ls;
  int d = t % DIN; t /= DIN;
  int b = t % Bn; int mod = t / Bn;
  const float* Y = (mod ? y_e : y_r) + (size_t)b * Kk * DIN * Ls;
  int tt = (s % Ww) * Hh + (s / Ww);  // col-major scan index for spatial s
  float v = Y[((size_t)(0 * DIN + d)) * Ls + s]
          + Y[((size_t)(2 * DIN + d)) * Ls + (Ls - 1 - s)]
          + Y[((size_t)(1 * DIN + d)) * Ls + tt]
          + Y[((size_t)(3 * DIN + d)) * Ls + (Ls - 1 - tt)];
  (mod ? ym_e : ym_r)[((size_t)b * DIN + d) * Ls + s] = v;
}

// ---------------- Kernel 6: spatial mean (squeeze) ----------------
__global__ void k_sq(const float* __restrict__ act_r, const float* __restrict__ act_e,
                     float* __restrict__ sq) {
  int idx = blockIdx.x * blockDim.x + threadIdx.x;
  const int total = 2 * Bn * DIN;
  if (idx >= total) return;
  int d = idx % DIN; int t = idx / DIN;
  int b = t % Bn; int mod = t / Bn;
  const float* A = (mod ? act_e : act_r) + ((size_t)b * DIN + d) * Ls;
  float s = 0.f;
  for (int l = 0; l < Ls; ++l) s += A[l];
  sq[(size_t)(mod * Bn + b) * DIN + d] = s / (float)Ls;
}

// ---------------- Kernel 7: SE excitation (192 -> 12 -> 192) ----------------
__global__ void k_se(const float* __restrict__ sq,
                     const float* __restrict__ f1w1, const float* __restrict__ f1w2,
                     const float* __restrict__ f2w1, const float* __restrict__ f2w2,
                     float* __restrict__ exc) {
  int z = blockIdx.x;           // mod*Bn + b
  int mod = z / Bn, b = z % Bn;
  const float* w1 = mod ? f2w1 : f1w1;
  const float* w2 = mod ? f2w2 : f1w2;
  const float* s = sq + (size_t)(mod * Bn + b) * DIN;
  __shared__ float hid[HID];
  int tid = threadIdx.x;
  if (tid < HID) {
    float a = 0.f;
    for (int dd = 0; dd < DIN; ++dd) a += w1[tid * DIN + dd] * s[dd];
    hid[tid] = silu_f(a);
  }
  __syncthreads();
  float o = 0.f;
#pragma unroll
  for (int hh = 0; hh < HID; ++hh) o += w2[tid * HID + hh] * hid[hh];
  exc[(size_t)(mod * Bn + b) * DIN + tid] = sigmoid_f(o);
}

// --------- Kernel 8: LayerNorm2d + cross SE scale + bf16 concat ---------
__global__ void k_ln(const float* __restrict__ ym_r, const float* __restrict__ ym_e,
                     const float* __restrict__ g1, const float* __restrict__ b1,
                     const float* __restrict__ g2, const float* __restrict__ b2,
                     const float* __restrict__ exc, unsigned short* __restrict__ ycat) {
  int idx = blockIdx.x * blockDim.x + threadIdx.x;
  const int total = 2 * Bn * Ls;
  if (idx >= total) return;
  int s = idx % Ls; int t = idx / Ls;
  int b = t % Bn; int mod = t / Bn;
  const float* Y = (mod ? ym_e : ym_r) + (size_t)b * DIN * Ls + s;
  float mu = 0.f;
  for (int d = 0; d < DIN; ++d) mu += Y[(size_t)d * Ls];
  mu /= (float)DIN;
  float var = 0.f;
  for (int d = 0; d < DIN; ++d) {
    float dd = Y[(size_t)d * Ls] - mu;
    var += dd * dd;
  }
  var /= (float)DIN;
  float inv = rsqrtf(var + 1e-5f);
  const float* g  = mod ? g2 : g1;
  const float* bb = mod ? b2 : b1;
  const float* ex = exc + (size_t)((1 - mod) * Bn + b) * DIN;  // cross scaling
  unsigned short* O = ycat + ((size_t)b * (2 * DIN) + (size_t)mod * DIN) * Ls + s;
  for (int d = 0; d < DIN; ++d) {
    float v = ((Y[(size_t)d * Ls] - mu) * inv * g[d] + bb[d]) * ex[d];
    O[(size_t)d * Ls] = bfbits(v);
  }
}

// ---------------- Kernel 9: out_proj GEMM (bf16 WMMA) ----------------
// out[b,o,l] = sum_c ow[o,c] * ycat[b,c,l]; M=96, N=1600, K=384
__global__ __launch_bounds__(32)
void k_outproj(const unsigned short* __restrict__ ycat,
               const float* __restrict__ ow, float* __restrict__ out) {
  const int lane = threadIdx.x;
  const int tn = blockIdx.x, tm = blockIdx.y, b = blockIdx.z;
  v8f acc = {};
  for (int kc = 0; kc < 2 * DIN; kc += 32) {
    acc = wmma_step(lane,
        [&](int m, int kk) { return tobf(ow[(tm * 16 + m) * (2 * DIN) + kc + kk]); },
        [&](int kk, int n) {
          return rawbf(ycat[((size_t)b * (2 * DIN) + kc + kk) * Ls + tn * 16 + n]);
        },
        acc);
  }
  const int nn = lane & 15, mb = ((lane >> 4) & 1) * 8;
#pragma unroll
  for (int v = 0; v < 8; ++v)
    out[((size_t)b * Cc + tm * 16 + mb + v) * Ls + tn * 16 + nn] = acc[v];
}

}  // namespace

extern "C" void kernel_launch(void* const* d_in, const int* in_sizes, int n_in,
                              void* d_out, int out_size, void* d_ws, size_t ws_size,
                              hipStream_t stream) {
  (void)in_sizes; (void)n_in; (void)out_size; (void)ws_size;

  const float* x_rgb      = (const float*)d_in[0];
  const float* x_e        = (const float*)d_in[1];
  const float* in_proj_w  = (const float*)d_in[2];
  const float* in_projm_w = (const float*)d_in[3];
  const float* conv_w     = (const float*)d_in[4];
  const float* conv_b     = (const float*)d_in[5];
  const float* xproj1_w   = (const float*)d_in[6];
  const float* xproj2_w   = (const float*)d_in[7];
  const float* dtproj1_w  = (const float*)d_in[8];
  const float* dtproj1_b  = (const float*)d_in[9];
  const float* dtproj2_w  = (const float*)d_in[10];
  const float* dtproj2_b  = (const float*)d_in[11];
  const float* A_log_1    = (const float*)d_in[12];
  const float* A_log_2    = (const float*)d_in[13];
  const float* D_1        = (const float*)d_in[14];
  const float* D_2        = (const float*)d_in[15];
  const float* norm1_g    = (const float*)d_in[16];
  const float* norm1_b    = (const float*)d_in[17];
  const float* norm2_g    = (const float*)d_in[18];
  const float* norm2_b    = (const float*)d_in[19];
  const float* fc1_w1     = (const float*)d_in[20];
  const float* fc1_w2     = (const float*)d_in[21];
  const float* fc2_w1     = (const float*)d_in[22];
  const float* fc2_w2     = (const float*)d_in[23];
  const float* out_proj_w = (const float*)d_in[24];
  float* out = (float*)d_out;

  // Workspace carve-up (all buffers fully overwritten every call; ~42 MB)
  float* ws = (float*)d_ws;
  size_t off = 0;
  auto carve = [&](size_t n) { float* p = ws + off; off += n; return p; };
  float* pre_r = carve((size_t)Bn * DIN * Ls);
  float* pre_e = carve((size_t)Bn * DIN * Ls);
  float* act_r = carve((size_t)Bn * DIN * Ls);
  float* act_e = carve((size_t)Bn * DIN * Ls);
  float* T_r   = carve((size_t)Bn * Kk * CPP * Ls);
  float* T_e   = carve((size_t)Bn * Kk * CPP * Ls);
  float* y_r   = carve((size_t)Bn * Kk * DIN * Ls);
  float* y_e   = carve((size_t)Bn * Kk * DIN * Ls);
  float* ym_r  = carve((size_t)Bn * DIN * Ls);
  float* ym_e  = carve((size_t)Bn * DIN * Ls);
  unsigned short* ycat = (unsigned short*)carve((size_t)Bn * 2 * DIN * Ls / 2);
  float* sqb   = carve((size_t)2 * Bn * DIN);
  float* excb  = carve((size_t)2 * Bn * DIN);

  // 1) in_proj (WMMA bf16): grid = (L/16, DIN/16, B*2 modalities)
  k_inproj<<<dim3(Ls / 16, DIN / 16, Bn * 2), dim3(32), 0, stream>>>(
      x_rgb, x_e, in_proj_w, in_projm_w, pre_r, pre_e);

  // 2) depthwise conv + SiLU
  {
    int total = 2 * Bn * DIN * Ls;
    k_conv<<<(total + 255) / 256, 256, 0, stream>>>(pre_r, pre_e, conv_w, conv_b,
                                                    act_r, act_e);
  }

  // 3) x_proj (WMMA bf16) over cross-scan views: grid = (L/16, 48/16, 2*B*K)
  k_xproj<<<dim3(Ls / 16, CPP / 16, 2 * Bn * Kk), dim3(32), 0, stream>>>(
      act_r, act_e, xproj1_w, xproj2_w, T_r, T_e);

  // 4) selective scan (cross-attended C and D)
  {
    int total = 2 * Bn * Kk * DIN;  // 3072 rows
    k_scan<<<(total + 127) / 128, 128, 0, stream>>>(
        act_r, act_e, T_r, T_e, A_log_1, A_log_2, dtproj1_w, dtproj1_b,
        dtproj2_w, dtproj2_b, D_1, D_2, y_r, y_e);
  }

  // 5) cross_merge
  {
    int total = 2 * Bn * DIN * Ls;
    k_merge<<<(total + 255) / 256, 256, 0, stream>>>(y_r, y_e, ym_r, ym_e);
  }

  // 6) squeeze (spatial mean)
  {
    int total = 2 * Bn * DIN;
    k_sq<<<(total + 255) / 256, 256, 0, stream>>>(act_r, act_e, sqb);
  }

  // 7) excitation MLPs
  k_se<<<2 * Bn, DIN, 0, stream>>>(sqb, fc1_w1, fc1_w2, fc2_w1, fc2_w2, excb);

  // 8) LayerNorm2d + cross SE scale + bf16 concat
  {
    int total = 2 * Bn * Ls;
    k_ln<<<(total + 255) / 256, 256, 0, stream>>>(ym_r, ym_e, norm1_g, norm1_b,
                                                  norm2_g, norm2_b, excb, ycat);
  }

  // 9) out_proj (WMMA bf16): grid = (L/16, C/16, B)
  k_outproj<<<dim3(Ls / 16, Cc / 16, Bn), dim3(32), 0, stream>>>(ycat, out_proj_w,
                                                                 out);
}
